// StackedEncoderModel_27178553049308
// MI455X (gfx1250) — compile-verified, matrix operand order checked by reading, not run on hardware
//
#include <hip/hip_runtime.h>

// ===========================================================================
// LRU stacked encoder for MI455X (gfx1250).
// All heavy math mapped onto v_wmma_f32_16x16x32_bf16 (f32 accumulate).
// Global->LDS staging uses CDNA5 async loads (GLOBAL_LOAD_ASYNC_TO_LDS_B128,
// ASYNCcnt-tracked) instead of a VGPR round trip.
// Complex algebra flattened into real bf16 GEMMs; recurrence solved with a
// chunked (64x64) parallel scan using closed-form powers of lambda.
// ===========================================================================

constexpr int T_SEQ   = 4096;
constexpr int D_M     = 2048;
constexpr int D_H     = 1024;
constexpr int N_LAYER = 4;

constexpr int NCHUNK = 64;
constexpr int CHLEN  = T_SEQ / NCHUNK;   // 64
constexpr int NPART  = 32;
constexpr int ROWS_PER_PART = T_SEQ / NPART;  // 128

typedef __bf16 bf16;
typedef __attribute__((ext_vector_type(16))) __bf16 v16bf;
typedef __attribute__((ext_vector_type(8)))  float  v8f;

union FragBF { v16bf v; uint4 q[2]; };

// ---------------------------------------------------------------------------
// WMMA GEMM:  C[M,N] (f32) = A[M,K](bf16 row-major) * B[N,K](bf16 row-major)^T
//             (+ optional per-column bias)
// Block: 256 threads = 8 waves. Tile 128x128, K-step 32.
// Wave grid 2(M) x 4(N); each wave computes 64x32 = 4x2 WMMA tiles.
// ---------------------------------------------------------------------------
constexpr int BM = 128, BN = 128, BK = 32;
constexpr int LDSP = 40;   // LDS row pitch in bf16 (80 B, padded from 64 B)

__global__ __launch_bounds__(256)
void gemm_bf16_wmma(const bf16* __restrict__ A, const bf16* __restrict__ B,
                    float* __restrict__ C, const float* __restrict__ bias,
                    int M, int N, int K)
{
  __shared__ bf16 smA[2][BM * LDSP];
  __shared__ bf16 smB[2][BN * LDSP];

  const int tid  = threadIdx.x;
  const int lane = tid & 31;
  const int wave = tid >> 5;
  const int wm   = wave & 1;          // 2 wave rows (64 M each)
  const int wn   = wave >> 1;         // 4 wave cols (32 N each)
  const int lr   = lane & 15;         // row / col within 16
  const int lk   = (lane >> 4) * 8;   // K sub-offset per ISA A/B layout

  const int bm = blockIdx.y * BM;
  const int bn = blockIdx.x * BN;

  // global->LDS staging: 128 rows x 64B per matrix; 2 x 16B chunks per thread,
  // moved with CDNA5 async-to-LDS loads (no VGPR staging).
  const int r0 = tid >> 2,        c0 = (tid & 3) * 8;
  const int r1 = (tid >> 2) + 64, c1 = c0;

  auto issue_async = [&](int kt, int buf) {
    const int k0 = kt * BK;
    const unsigned long long gA0 =
        (unsigned long long)(size_t)(A + (size_t)(bm + r0) * K + k0 + c0);
    const unsigned long long gA1 =
        (unsigned long long)(size_t)(A + (size_t)(bm + r1) * K + k0 + c1);
    const unsigned long long gB0 =
        (unsigned long long)(size_t)(B + (size_t)(bn + r0) * K + k0 + c0);
    const unsigned long long gB1 =
        (unsigned long long)(size_t)(B + (size_t)(bn + r1) * K + k0 + c1);
    const unsigned lA0 = (unsigned)(size_t)&smA[buf][r0 * LDSP + c0];
    const unsigned lA1 = (unsigned)(size_t)&smA[buf][r1 * LDSP + c1];
    const unsigned lB0 = (unsigned)(size_t)&smB[buf][r0 * LDSP + c0];
    const unsigned lB1 = (unsigned)(size_t)&smB[buf][r1 * LDSP + c1];
    asm volatile("global_load_async_to_lds_b128 %0, %1, off"
                 :: "v"(lA0), "v"(gA0) : "memory");
    asm volatile("global_load_async_to_lds_b128 %0, %1, off"
                 :: "v"(lA1), "v"(gA1) : "memory");
    asm volatile("global_load_async_to_lds_b128 %0, %1, off"
                 :: "v"(lB0), "v"(gB0) : "memory");
    asm volatile("global_load_async_to_lds_b128 %0, %1, off"
                 :: "v"(lB1), "v"(gB1) : "memory");
  };

  v8f acc[4][2];
#pragma unroll
  for (int mt = 0; mt < 4; ++mt)
#pragma unroll
    for (int nt = 0; nt < 2; ++nt)
#pragma unroll
      for (int r = 0; r < 8; ++r) acc[mt][nt][r] = 0.0f;

  auto compute = [&](int buf) {
    FragBF fa[4], fb[2];
#pragma unroll
    for (int mt = 0; mt < 4; ++mt) {
      const bf16* p = &smA[buf][(wm * 64 + mt * 16 + lr) * LDSP + lk];
      fa[mt].q[0] = *(const uint4*)p;          // K = lk .. lk+7
      fa[mt].q[1] = *(const uint4*)(p + 16);   // K = 16+lk .. 16+lk+7
    }
#pragma unroll
    for (int nt = 0; nt < 2; ++nt) {
      const bf16* p = &smB[buf][(wn * 32 + nt * 16 + lr) * LDSP + lk];
      fb[nt].q[0] = *(const uint4*)p;
      fb[nt].q[1] = *(const uint4*)(p + 16);
    }
#pragma unroll
    for (int mt = 0; mt < 4; ++mt)
#pragma unroll
      for (int nt = 0; nt < 2; ++nt)
        acc[mt][nt] = __builtin_amdgcn_wmma_f32_16x16x32_bf16(
            false, fa[mt].v, false, fb[nt].v, (short)0, acc[mt][nt],
            false, false);
  };

  const int nk = K / BK;

  issue_async(0, 0);
  asm volatile("s_wait_asynccnt 0x0" ::: "memory");
  __syncthreads();

  for (int kt = 0; kt < nk; ++kt) {
    if (kt + 1 < nk) {
      // prefetch-to-LDS of the next tile overlaps with this tile's WMMAs
      issue_async(kt + 1, (kt + 1) & 1);
      if (kt + 2 < nk) {  // gfx1250 global_prefetch_b8 two tiles ahead (L2 warm)
        __builtin_prefetch(A + (size_t)(bm + (tid >> 1)) * K + (kt + 2) * BK, 0, 1);
        __builtin_prefetch(B + (size_t)(bn + (tid >> 1)) * K + (kt + 2) * BK, 0, 1);
      }
    }
    compute(kt & 1);
    // my async writes for the next tile are LDS-visible, then rendezvous
    asm volatile("s_wait_asynccnt 0x0" ::: "memory");
    __syncthreads();
  }

  // Epilogue: C 16x16 f32 layout -> VGPR r holds M = r + 8*(lane>=16), N = lane&15
#pragma unroll
  for (int mt = 0; mt < 4; ++mt) {
    const int mrow = bm + wm * 64 + mt * 16 + (lane >> 4) * 8;
#pragma unroll
    for (int nt = 0; nt < 2; ++nt) {
      const int n = bn + wn * 32 + nt * 16 + lr;
      const float bv = bias ? bias[n] : 0.0f;
#pragma unroll
      for (int r = 0; r < 8; ++r)
        C[(size_t)(mrow + r) * N + n] = acc[mt][nt][r] + bv;
    }
  }
}

// ---------------------------------------------------------------------------
// BatchNorm over time axis (training-mode, biased var), deterministic tree sum
// ---------------------------------------------------------------------------
__global__ void bn_partial_kernel(const float* __restrict__ x, float* __restrict__ part)
{
  const int c = blockIdx.x * blockDim.x + threadIdx.x;  // channel
  const int p = blockIdx.y;                             // row chunk
  float s = 0.f, q = 0.f;
  const int t0 = p * ROWS_PER_PART;
  for (int i = 0; i < ROWS_PER_PART; ++i) {
    const float v = x[(size_t)(t0 + i) * D_M + c];
    s += v; q += v * v;
  }
  part[(size_t)p * D_M + c] = s;
  part[(size_t)(NPART + p) * D_M + c] = q;
}

__global__ void bn_reduce_kernel(const float* __restrict__ part, float* __restrict__ sums)
{
  const int c = blockIdx.x * blockDim.x + threadIdx.x;
  float s = 0.f, q = 0.f;
  for (int p = 0; p < NPART; ++p) {
    s += part[(size_t)p * D_M + c];
    q += part[(size_t)(NPART + p) * D_M + c];
  }
  sums[c] = s;
  sums[D_M + c] = q;
}

__global__ void bn_apply_kernel(const float* __restrict__ x, const float* __restrict__ sums,
                                const float* __restrict__ nw, const float* __restrict__ nb,
                                float* __restrict__ xnf, bf16* __restrict__ xnb)
{
  const size_t i = (size_t)blockIdx.x * blockDim.x + threadIdx.x;
  const int m = (int)(i % D_M);
  const float mu  = sums[m] * (1.0f / T_SEQ);
  const float var = sums[D_M + m] * (1.0f / T_SEQ) - mu * mu;
  const float r = rsqrtf(var + 1e-5f);
  const float v = (x[i] - mu) * r * nw[m] + nb[m];
  xnf[i] = v;
  xnb[i] = (bf16)v;
}

// ---------------------------------------------------------------------------
// Weight repack / conversion
// ---------------------------------------------------------------------------
__global__ void cvt_f32_bf16_kernel(const float* __restrict__ in, bf16* __restrict__ out)
{
  const size_t i = (size_t)blockIdx.x * blockDim.x + threadIdx.x;
  out[i] = (bf16)in[i];
}

// Wb[2*DH, DM]: rows [0,DH) = exp(gl)*B_re, rows [DH,2DH) = exp(gl)*B_im
__global__ void build_wb_kernel(const float* __restrict__ br, const float* __restrict__ bi,
                                const float* __restrict__ gl, bf16* __restrict__ w)
{
  const size_t i = (size_t)blockIdx.x * blockDim.x + threadIdx.x;  // over 2*DH*DM
  const int r = (int)(i / D_M);
  const int m = (int)(i % D_M);
  const float s = __expf(gl[r & (D_H - 1)]);
  const float v = (r < D_H) ? br[(size_t)r * D_M + m]
                            : bi[(size_t)(r - D_H) * D_M + m];
  w[i] = (bf16)(v * s);
}

// Wc[DM, 2*DH]: cols [0,DH) = C_re, cols [DH,2DH) = -C_im  (real part of h @ C^T)
__global__ void build_wc_kernel(const float* __restrict__ cr, const float* __restrict__ ci,
                                bf16* __restrict__ w)
{
  const size_t i = (size_t)blockIdx.x * blockDim.x + threadIdx.x;  // over DM*2*DH
  const int m = (int)(i / (2 * D_H));
  const int k = (int)(i % (2 * D_H));
  const float v = (k < D_H) ? cr[(size_t)m * D_H + k]
                            : -ci[(size_t)m * D_H + (k - D_H)];
  w[i] = (bf16)v;
}

__global__ void pack_bias_kernel(const float* __restrict__ b1, const float* __restrict__ b2,
                                 float* __restrict__ bg)
{
  const int i = blockIdx.x * blockDim.x + threadIdx.x;
  bg[i] = (i < D_M) ? b1[i] : b2[i - D_M];
}

// ---------------------------------------------------------------------------
// Chunked parallel scan:  h_t = lambda*h_{t-1} + Bu_t   (complex, per channel)
// lambda = exp(-exp(nu) + i*exp(th));  lambda^k via expf/sincosf
// ---------------------------------------------------------------------------
__global__ void lru_scan_pass1(float* __restrict__ bu, const float* __restrict__ nu,
                               const float* __restrict__ th, float* __restrict__ carry)
{
  const int c = blockIdx.x * blockDim.x + threadIdx.x;  // channel
  const int j = blockIdx.y;                             // chunk
  const float e   = __expf(nu[c]);
  const float ang = __expf(th[c]);
  const float mag = __expf(-e);
  const float lre = mag * __cosf(ang);
  const float lim = mag * __sinf(ang);
  float hre = 0.f, him = 0.f;
  const int t0 = j * CHLEN;
  for (int k = 0; k < CHLEN; ++k) {
    const size_t base = (size_t)(t0 + k) * (2 * D_H);
    const float nre = lre * hre - lim * him + bu[base + c];
    const float nim = lre * him + lim * hre + bu[base + D_H + c];
    hre = nre; him = nim;
    bu[base + c]       = hre;
    bu[base + D_H + c] = him;
  }
  carry[(size_t)j * (2 * D_H) + c]       = hre;
  carry[(size_t)j * (2 * D_H) + D_H + c] = him;
}

__global__ void lru_scan_pass2(const float* __restrict__ carry, const float* __restrict__ nu,
                               const float* __restrict__ th, float* __restrict__ prefix)
{
  const int c = blockIdx.x * blockDim.x + threadIdx.x;
  const float e   = __expf(nu[c]);
  const float ang = __expf(th[c]);
  const float magC = __expf(-(float)CHLEN * e);
  float sC, cC;
  __sincosf((float)CHLEN * ang, &sC, &cC);
  const float Lre = magC * cC, Lim = magC * sC;  // lambda^CHLEN
  float pre = 0.f, pim = 0.f;
  for (int j = 0; j < NCHUNK; ++j) {
    const size_t base = (size_t)j * (2 * D_H);
    prefix[base + c]       = pre;
    prefix[base + D_H + c] = pim;
    const float nre = Lre * pre - Lim * pim + carry[base + c];
    const float nim = Lre * pim + Lim * pre + carry[base + D_H + c];
    pre = nre; pim = nim;
  }
}

__global__ void lru_scan_pass3(const float* __restrict__ hloc, const float* __restrict__ prefix,
                               const float* __restrict__ nu, const float* __restrict__ th,
                               bf16* __restrict__ hcat)
{
  const int c = blockIdx.x * blockDim.x + threadIdx.x;
  const int j = blockIdx.y;
  const float e   = __expf(nu[c]);
  const float ang = __expf(th[c]);
  const float mag = __expf(-e);
  const float lre = mag * __cosf(ang);
  const float lim = mag * __sinf(ang);
  const float pre = prefix[(size_t)j * (2 * D_H) + c];
  const float pim = prefix[(size_t)j * (2 * D_H) + D_H + c];
  float pwr = lre, pwi = lim;  // lambda^(k+1)
  const int t0 = j * CHLEN;
  for (int k = 0; k < CHLEN; ++k) {
    const size_t base = (size_t)(t0 + k) * (2 * D_H);
    const float cre = pwr * pre - pwi * pim;
    const float cim = pwr * pim + pwi * pre;
    hcat[base + c]       = (bf16)(hloc[base + c] + cre);
    hcat[base + D_H + c] = (bf16)(hloc[base + D_H + c] + cim);
    const float nr = pwr * lre - pwi * lim;
    pwi = pwr * lim + pwi * lre;
    pwr = nr;
  }
}

// ---------------------------------------------------------------------------
// Epilogues
// ---------------------------------------------------------------------------
__global__ void ypost_kernel(const float* __restrict__ y0, const float* __restrict__ xn,
                             const float* __restrict__ d, bf16* __restrict__ out)
{
  const size_t i = (size_t)blockIdx.x * blockDim.x + threadIdx.x;
  const int m = (int)(i % D_M);
  const float v = y0[i] + xn[i] * d[m];
  const float g = 0.5f * v * (1.0f + erff(v * 0.70710678118654752f));  // exact GELU
  out[i] = (bf16)g;
}

__global__ void glu_residual_kernel(const float* __restrict__ g, float* __restrict__ carry)
{
  const size_t i = (size_t)blockIdx.x * blockDim.x + threadIdx.x;
  const size_t t = i / D_M;
  const int    m = (int)(i % D_M);
  const float a = g[t * (2 * D_M) + m];
  const float b = g[t * (2 * D_M) + D_M + m];
  carry[i] += a * (1.0f / (1.0f + __expf(-b)));
}

// ---------------------------------------------------------------------------
// Host orchestration
// ---------------------------------------------------------------------------
extern "C" void kernel_launch(void* const* d_in, const int* in_sizes, int n_in,
                              void* d_out, int out_size, void* d_ws, size_t ws_size,
                              hipStream_t stream)
{
  (void)in_sizes; (void)n_in; (void)out_size; (void)ws_size;

  const float* x      = (const float*)d_in[0];
  const float* enc_w  = (const float*)d_in[1];
  const float* enc_b  = (const float*)d_in[2];
  const float* nu_log = (const float*)d_in[3];
  const float* th_log = (const float*)d_in[4];
  const float* gl_log = (const float*)d_in[5];
  const float* B_re   = (const float*)d_in[6];
  const float* B_im   = (const float*)d_in[7];
  const float* C_re   = (const float*)d_in[8];
  const float* C_im   = (const float*)d_in[9];
  const float* Dvec   = (const float*)d_in[10];
  const float* norm_w = (const float*)d_in[11];
  const float* norm_b = (const float*)d_in[12];
  const float* w1     = (const float*)d_in[13];
  const float* b1     = (const float*)d_in[14];
  const float* w2     = (const float*)d_in[15];
  const float* b2     = (const float*)d_in[16];

  float* out = (float*)d_out;  // doubles as the residual carry [T, DM]

  // workspace carve-up
  char* ws = (char*)d_ws;
  size_t off = 0;
  auto take = [&](size_t bytes) -> char* {
    char* p = ws + off;
    off = (off + bytes + 255) & ~(size_t)255;
    return p;
  };
  bf16*  ABF   = (bf16*) take((size_t)T_SEQ * D_M * sizeof(bf16));      // bf16 activations
  bf16*  WBF   = (bf16*) take((size_t)2 * D_M * D_M * sizeof(bf16));    // bf16 weights (max 2DMxDM)
  float* XN    = (float*)take((size_t)T_SEQ * D_M * sizeof(float));     // xn f32
  float* TMP   = (float*)take((size_t)T_SEQ * 2 * D_M * sizeof(float)); // f32 GEMM output
  float* CARRY = (float*)take((size_t)NCHUNK * 2 * D_H * sizeof(float));
  float* PREFX = (float*)take((size_t)NCHUNK * 2 * D_H * sizeof(float));
  float* PART  = (float*)take((size_t)2 * NPART * D_M * sizeof(float));
  float* SUMS  = (float*)take((size_t)2 * D_M * sizeof(float));
  float* BGLU  = (float*)take((size_t)2 * D_M * sizeof(float));

  const dim3 blk(256);
  const dim3 gEW((D_M * D_M) / 256);
  const dim3 gTD((T_SEQ * D_M) / 256);
  const dim3 gemmN1(D_M / BN, T_SEQ / BM);      // N = 2048
  const dim3 gemmN2((2 * D_M) / BN, T_SEQ / BM);// N = 4096
  const dim3 gScan(D_H / 256, NCHUNK);

  // encoder: out = x @ enc_w^T + enc_b
  cvt_f32_bf16_kernel<<<gTD, blk, 0, stream>>>(x, ABF);
  cvt_f32_bf16_kernel<<<gEW, blk, 0, stream>>>(enc_w, WBF);
  gemm_bf16_wmma<<<gemmN1, blk, 0, stream>>>(ABF, WBF, out, enc_b, T_SEQ, D_M, D_M);

  for (int l = 0; l < N_LAYER; ++l) {
    const size_t lDH  = (size_t)l * D_H;
    const size_t lDM  = (size_t)l * D_M;
    const size_t lBW  = (size_t)l * D_H * D_M;
    const size_t lCW  = (size_t)l * D_M * D_H;
    const size_t lWW  = (size_t)l * D_M * D_M;

    // BatchNorm (training mode over T)
    bn_partial_kernel<<<dim3(D_M / 256, NPART), blk, 0, stream>>>(out, PART);
    bn_reduce_kernel<<<dim3(D_M / 256), blk, 0, stream>>>(PART, SUMS);
    bn_apply_kernel<<<gTD, blk, 0, stream>>>(out, SUMS, norm_w + lDM, norm_b + lDM, XN, ABF);

    // Bu = xn @ [gamma*B_re ; gamma*B_im]^T   (one GEMM, N = 2*DH)
    build_wb_kernel<<<dim3((2 * D_H * D_M) / 256), blk, 0, stream>>>(B_re + lBW, B_im + lBW,
                                                                     gl_log + lDH, WBF);
    gemm_bf16_wmma<<<gemmN1, blk, 0, stream>>>(ABF, WBF, TMP, nullptr, T_SEQ, 2 * D_H, D_M);

    // parallel scan -> Hcat bf16 = [h_re | h_im]
    lru_scan_pass1<<<gScan, blk, 0, stream>>>(TMP, nu_log + lDH, th_log + lDH, CARRY);
    lru_scan_pass2<<<dim3(D_H / 256), blk, 0, stream>>>(CARRY, nu_log + lDH, th_log + lDH, PREFX);
    lru_scan_pass3<<<gScan, blk, 0, stream>>>(TMP, PREFX, nu_log + lDH, th_log + lDH, ABF);

    // y0 = Hcat @ [C_re | -C_im]^T
    build_wc_kernel<<<dim3((D_M * 2 * D_H) / 256), blk, 0, stream>>>(C_re + lCW, C_im + lCW, WBF);
    gemm_bf16_wmma<<<gemmN1, blk, 0, stream>>>(ABF, WBF, TMP, nullptr, T_SEQ, D_M, 2 * D_H);

    // y = gelu(y0 + xn*D)  (bf16 for the GLU GEMM)
    ypost_kernel<<<gTD, blk, 0, stream>>>(TMP, XN, Dvec + lDM, ABF);

    // g = y @ [w1 ; w2]^T + [b1 ; b2]   (one GEMM, N = 2*DM)
    cvt_f32_bf16_kernel<<<gEW, blk, 0, stream>>>(w1 + lWW, WBF);
    cvt_f32_bf16_kernel<<<gEW, blk, 0, stream>>>(w2 + lWW, WBF + (size_t)D_M * D_M);
    pack_bias_kernel<<<dim3((2 * D_M) / 256), blk, 0, stream>>>(b1 + lDM, b2 + lDM, BGLU);
    gemm_bf16_wmma<<<gemmN2, blk, 0, stream>>>(ABF, WBF, TMP, BGLU, T_SEQ, 2 * D_M, D_M);

    // carry += g1 * sigmoid(g2)
    glu_residual_kernel<<<gTD, blk, 0, stream>>>(TMP, out);
  }
}
